// LSTMNetwork_13065290515157
// MI455X (gfx1250) — matrix operand order repaired and checked by reading, not compiled, not run on hardware
//
#include <hip/hip_runtime.h>
#include <math.h>

// LSTM: B=256, T=2048, F=64, H=512, C=128
#define B_  256
#define T_  2048
#define F_  64
#define H_  512
#define C_  128

#define BT      32              // batch rows per tile (2 WMMA M-tiles)
#define NBT     (B_/BT)         // 8 batch tiles
#define HC      32              // h-columns per compute WG
#define NCC     (H_/HC)         // 16 compute WGs per tile (+1 decoder)
#define THREADS 256             // 8 waves of 32

// ---- dynamic LDS layout (ushort offsets / byte offsets) ----
#define OFF_WB 0                // Whh^T chunk (compute) / Wd^T (decoder): 65536 us = 128KB
#define OFF_WI 65536            // Wih^T chunk: 8192 us = 16KB
#define OFF_H  73728            // h A-fragments: 16384 us = 32KB
#define OFF_X  90112            // x A-fragments: 2048 us = 4KB
#define FB_BYTES 184320         // float region start
#define SMEM_BYTES 205312       // + gates 16KB + cell 4KB + bias 512B

typedef __attribute__((ext_vector_type(16))) __bf16 v16bf;
typedef __attribute__((ext_vector_type(8)))  float  v8f;

static __device__ __forceinline__ unsigned short f2bf(float f) {
  unsigned int u = __float_as_uint(f);
  u += 0x7fffu + ((u >> 16) & 1u);          // round-to-nearest-even
  return (unsigned short)(u >> 16);
}

__global__ void init_counters_k(int* c) {
  if (threadIdx.x < 64) c[threadIdx.x] = 0;
}

// Fragment layouts used here (contiguous k order, same for A and B):
//  A 16x32 bf16:  lane = m%16 + 16*(kk>>4),     elem e = kk & 15
//  B 32x16 bf16:  lane = n%16 + 16*(kk>>4),     elem e = kk & 15
//  C/D 16x16 f32: lane = n + 16*(m>=8 half),    vgpr r -> m = r + 8*(lane>>4)

__launch_bounds__(THREADS)
__global__ void lstm_wmma_persistent(
    const float* __restrict__ x,   const float* __restrict__ Wih,
    const float* __restrict__ Whh, const float* __restrict__ bih,
    const float* __restrict__ bhh, const float* __restrict__ Wd,
    const float* __restrict__ bd,  float* __restrict__ out,
    int* __restrict__ ctrs, unsigned short* __restrict__ hbuf)
{
  extern __shared__ unsigned char smem[];
  unsigned short* sWB = (unsigned short*)smem + OFF_WB;
  unsigned short* sWI = (unsigned short*)smem + OFF_WI;
  unsigned short* sH  = (unsigned short*)smem + OFF_H;
  unsigned short* sX  = (unsigned short*)smem + OFF_X;
  float* sG    = (float*)(smem + FB_BYTES);   // gates / logits [32][128]
  float* sC    = sG + 4096;                   // cell state [32][32]
  float* sBias = sC + 1024;                   // [128]

  const int tid  = threadIdx.x;
  const int wv   = tid >> 5;
  const int lane = tid & 31;
  const int cc   = blockIdx.x;      // 0..NCC-1 compute; ==NCC decoder
  const int bt   = blockIdx.y;      // batch tile
  const int rb   = bt * BT;
  int* wctr = ctrs + bt;            // h-version write counter (+NCC per step)
  int* dctr = ctrs + NBT + bt;      // decoder read counter (+1 per step)

  if (cc < NCC) {
    // ================= compute workgroup =================
    // Preload Whh^T chunk into B-fragment swizzle (one-time)
    for (int c = tid; c < 8192; c += THREADS) {
      int e0 = (c & 1) * 8, ln = (c >> 1) & 31, nt = (c >> 6) & 7, kb = (c >> 9) & 15;
      int k  = kb * 32 + (ln >> 4) * 16 + e0;
      int lgc = nt * 16 + (ln & 15);
      int gc  = (lgc >> 5) * H_ + cc * HC + (lgc & 31);   // global gate col (i,f,g,o blocks)
      const float* s = Whh + (size_t)gc * H_ + k;
      unsigned short* d = sWB + (((kb * 8 + nt) * 32 + ln) << 4) + e0;
      #pragma unroll
      for (int i = 0; i < 8; ++i) d[i] = f2bf(s[i]);
    }
    // Preload Wih^T chunk
    for (int c = tid; c < 1024; c += THREADS) {
      int e0 = (c & 1) * 8, ln = (c >> 1) & 31, nt = (c >> 6) & 7, kb = (c >> 9) & 1;
      int k  = kb * 32 + (ln >> 4) * 16 + e0;
      int lgc = nt * 16 + (ln & 15);
      int gc  = (lgc >> 5) * H_ + cc * HC + (lgc & 31);
      const float* s = Wih + (size_t)gc * F_ + k;
      unsigned short* d = sWI + (((kb * 8 + nt) * 32 + ln) << 4) + e0;
      #pragma unroll
      for (int i = 0; i < 8; ++i) d[i] = f2bf(s[i]);
    }
    for (int i = tid; i < 128; i += THREADS) {
      int gc = (i >> 5) * H_ + cc * HC + (i & 31);
      sBias[i] = bih[gc] + bhh[gc];
    }
    for (int i = tid; i < BT * HC; i += THREADS) sC[i] = 0.f;
    for (int i = tid; i < BT * HC; i += THREADS) {   // h_0 = 0 (version 0, buffer 0)
      int row = i >> 5, j = i & 31;
      hbuf[(size_t)(rb + row) * H_ + cc * HC + j] = 0;
    }
    __threadfence();
    __syncthreads();
    if (tid == 0) __hip_atomic_fetch_add(wctr, 1, __ATOMIC_RELEASE, __HIP_MEMORY_SCOPE_AGENT);

    for (int t = 0; t < T_; ++t) {
      // wait: h version t fully written (also implies all peers finished step t-1)
      if (tid == 0) {
        int tgt = NCC * (t + 1);
        while (__hip_atomic_load(wctr, __ATOMIC_ACQUIRE, __HIP_MEMORY_SCOPE_AGENT) < tgt)
          __builtin_amdgcn_s_sleep(1);
      }
      __syncthreads();
      // copy h_t tile (32x512 bf16) into A-fragment swizzle
      const unsigned short* hs = hbuf + ((size_t)(t & 1) * B_ + rb) * H_;
      for (int c = tid; c < 2048; c += THREADS) {
        int e0 = (c & 1) * 8, ln = (c >> 1) & 31, kb = (c >> 6) & 15, mt = (c >> 10) & 1;
        int k  = kb * 32 + ((ln >> 4) << 4) + ((c & 1) << 3);
        int m  = mt * 16 + (ln & 15);
        *(uint4*)(sH + (((mt * 16 + kb) * 32 + ln) << 4) + e0) =
            *(const uint4*)(hs + (size_t)m * H_ + k);
      }
      // copy+convert x_t tile (32x64 f32 -> bf16 A-fragments)
      {
        int c = tid;
        int e0 = (c & 1) * 8, ln = (c >> 1) & 31, kb = (c >> 6) & 1, mt = (c >> 7) & 1;
        int k  = kb * 32 + ((ln >> 4) << 4) + ((c & 1) << 3);
        int m  = mt * 16 + (ln & 15);
        const float* s = x + ((size_t)(rb + m) * T_ + t) * F_ + k;
        float4 f0 = ((const float4*)s)[0];
        float4 f1 = ((const float4*)s)[1];
        unsigned short* d = sX + (((mt * 2 + kb) * 32 + ln) << 4) + e0;
        d[0]=f2bf(f0.x); d[1]=f2bf(f0.y); d[2]=f2bf(f0.z); d[3]=f2bf(f0.w);
        d[4]=f2bf(f1.x); d[5]=f2bf(f1.y); d[6]=f2bf(f1.z); d[7]=f2bf(f1.w);
        if (t + 1 < T_) __builtin_prefetch(s + F_, 0, 1);  // next timestep's x
      }
      __syncthreads();
      // gates GEMM: [32,512]x[512,128] + [32,64]x[64,128], f32 accumulate
      v8f acc0 = {}, acc1 = {};
      #pragma unroll
      for (int kb = 0; kb < 16; ++kb) {
        v16bf bb = *(const v16bf*)(sWB + (((kb * 8 + wv) * 32 + lane) << 4));
        v16bf a0 = *(const v16bf*)(sH + ((kb * 32 + lane) << 4));
        v16bf a1 = *(const v16bf*)(sH + (((16 + kb) * 32 + lane) << 4));
        acc0 = __builtin_amdgcn_wmma_f32_16x16x32_bf16(false, a0, false, bb, (short)0, acc0, false, false);
        acc1 = __builtin_amdgcn_wmma_f32_16x16x32_bf16(false, a1, false, bb, (short)0, acc1, false, false);
      }
      #pragma unroll
      for (int kb = 0; kb < 2; ++kb) {
        v16bf bb = *(const v16bf*)(sWI + (((kb * 8 + wv) * 32 + lane) << 4));
        v16bf a0 = *(const v16bf*)(sX + ((kb * 32 + lane) << 4));
        v16bf a1 = *(const v16bf*)(sX + (((2 + kb) * 32 + lane) << 4));
        acc0 = __builtin_amdgcn_wmma_f32_16x16x32_bf16(false, a0, false, bb, (short)0, acc0, false, false);
        acc1 = __builtin_amdgcn_wmma_f32_16x16x32_bf16(false, a1, false, bb, (short)0, acc1, false, false);
      }
      float bv = sBias[wv * 16 + (lane & 15)];
      int nn = lane & 15, mh = lane >> 4;
      #pragma unroll
      for (int r = 0; r < 8; ++r) {
        sG[(r + 8 * mh) * 128 + wv * 16 + nn]      = acc0[r] + bv;
        sG[(16 + r + 8 * mh) * 128 + wv * 16 + nn] = acc1[r] + bv;
      }
      __syncthreads();
      // before overwriting h buffer (t+1)&1 (holds version t-1): decoder must have read it
      if (tid == 0 && t >= 2) {
        int tgt = t - 1;
        while (__hip_atomic_load(dctr, __ATOMIC_ACQUIRE, __HIP_MEMORY_SCOPE_AGENT) < tgt)
          __builtin_amdgcn_s_sleep(1);
      }
      __syncthreads();
      // elementwise LSTM cell update (i,f,g,o blocks of 32 local cols)
      unsigned short* hd = hbuf + ((size_t)((t + 1) & 1) * B_ + rb) * H_ + cc * HC;
      #pragma unroll
      for (int it = 0; it < 4; ++it) {
        int cell = tid + it * THREADS;
        int row = cell >> 5, j = cell & 31;
        float gi = sG[row * 128 + j];
        float gf = sG[row * 128 + 32 + j];
        float gg = sG[row * 128 + 64 + j];
        float go = sG[row * 128 + 96 + j];
        float ii = 1.f / (1.f + __expf(-gi));
        float ff = 1.f / (1.f + __expf(-gf));
        float g2 = tanhf(gg);
        float oo = 1.f / (1.f + __expf(-go));
        float cn = ff * sC[cell] + ii * g2;
        sC[cell] = cn;
        hd[(size_t)row * H_ + j] = f2bf(oo * tanhf(cn));
      }
      __threadfence();
      __syncthreads();
      if (tid == 0) __hip_atomic_fetch_add(wctr, 1, __ATOMIC_RELEASE, __HIP_MEMORY_SCOPE_AGENT);
    }
  } else {
    // ================= decoder workgroup =================
    for (int c = tid; c < 8192; c += THREADS) {     // Wd^T -> B-fragment swizzle
      int e0 = (c & 1) * 8, ln = (c >> 1) & 31, nt = (c >> 6) & 7, kb = (c >> 9) & 15;
      int k  = kb * 32 + (ln >> 4) * 16 + e0;
      int n  = nt * 16 + (ln & 15);
      const float* s = Wd + (size_t)n * H_ + k;
      unsigned short* d = sWB + (((kb * 8 + nt) * 32 + ln) << 4) + e0;
      #pragma unroll
      for (int i = 0; i < 8; ++i) d[i] = f2bf(s[i]);
    }
    for (int i = tid; i < 128; i += THREADS) sBias[i] = bd[i];
    __syncthreads();

    const size_t orow = (size_t)T_ * C_;
    for (int t = 0; t < T_; ++t) {
      // out_t uses h version t+1
      if (tid == 0) {
        int tgt = NCC * (t + 2);
        while (__hip_atomic_load(wctr, __ATOMIC_ACQUIRE, __HIP_MEMORY_SCOPE_AGENT) < tgt)
          __builtin_amdgcn_s_sleep(1);
      }
      __syncthreads();
      // copy h_{t+1} with relu applied, into A-fragment swizzle
      const unsigned short* hs = hbuf + ((size_t)((t + 1) & 1) * B_ + rb) * H_;
      for (int c = tid; c < 2048; c += THREADS) {
        int e0 = (c & 1) * 8, ln = (c >> 1) & 31, kb = (c >> 6) & 15, mt = (c >> 10) & 1;
        int k  = kb * 32 + ((ln >> 4) << 4) + ((c & 1) << 3);
        int m  = mt * 16 + (ln & 15);
        uint4 v = *(const uint4*)(hs + (size_t)m * H_ + k);
        unsigned short* pv = (unsigned short*)&v;
        #pragma unroll
        for (int i = 0; i < 8; ++i) if (pv[i] & 0x8000u) pv[i] = 0;   // relu on bf16
        *(uint4*)(sH + (((mt * 16 + kb) * 32 + ln) << 4) + e0) = v;
      }
      __syncthreads();
      if (tid == 0) __hip_atomic_fetch_add(dctr, 1, __ATOMIC_RELEASE, __HIP_MEMORY_SCOPE_AGENT);
      // logits GEMM: [32,512]x[512,128]
      v8f acc0 = {}, acc1 = {};
      #pragma unroll
      for (int kb = 0; kb < 16; ++kb) {
        v16bf bb = *(const v16bf*)(sWB + (((kb * 8 + wv) * 32 + lane) << 4));
        v16bf a0 = *(const v16bf*)(sH + ((kb * 32 + lane) << 4));
        v16bf a1 = *(const v16bf*)(sH + (((16 + kb) * 32 + lane) << 4));
        acc0 = __builtin_amdgcn_wmma_f32_16x16x32_bf16(false, a0, false, bb, (short)0, acc0, false, false);
        acc1 = __builtin_amdgcn_wmma_f32_16x16x32_bf16(false, a1, false, bb, (short)0, acc1, false, false);
      }
      float bv = sBias[wv * 16 + (lane & 15)];
      int nn = lane & 15, mh = lane >> 4;
      #pragma unroll
      for (int r = 0; r < 8; ++r) {
        sG[(r + 8 * mh) * 128 + wv * 16 + nn]      = acc0[r] + bv;
        sG[(16 + r + 8 * mh) * 128 + wv * 16 + nn] = acc1[r] + bv;
      }
      __syncthreads();
      // softmax over C=128 per row; 4 rows per wave, 4 cols per lane
      #pragma unroll
      for (int rr = 0; rr < 4; ++rr) {
        int row = wv * 4 + rr;
        float4 v = *(const float4*)(sG + row * 128 + lane * 4);
        float mx = fmaxf(fmaxf(v.x, v.y), fmaxf(v.z, v.w));
        #pragma unroll
        for (int s = 16; s > 0; s >>= 1) mx = fmaxf(mx, __shfl_xor(mx, s, 32));
        v.x = __expf(v.x - mx); v.y = __expf(v.y - mx);
        v.z = __expf(v.z - mx); v.w = __expf(v.w - mx);
        float sm = v.x + v.y + v.z + v.w;
        #pragma unroll
        for (int s = 16; s > 0; s >>= 1) sm += __shfl_xor(sm, s, 32);
        float inv = 1.f / sm;
        v.x *= inv; v.y *= inv; v.z *= inv; v.w *= inv;
        *(float4*)(out + (size_t)(rb + row) * orow + (size_t)t * C_ + lane * 4) = v;
      }
      __syncthreads();
    }
  }
}

extern "C" void kernel_launch(void* const* d_in, const int* in_sizes, int n_in,
                              void* d_out, int out_size, void* d_ws, size_t ws_size,
                              hipStream_t stream) {
  (void)in_sizes; (void)n_in; (void)out_size; (void)ws_size;
  const float* x   = (const float*)d_in[0];
  const float* Wih = (const float*)d_in[1];
  const float* Whh = (const float*)d_in[2];
  const float* bih = (const float*)d_in[3];
  const float* bhh = (const float*)d_in[4];
  const float* Wd  = (const float*)d_in[5];
  const float* bd  = (const float*)d_in[6];
  float* out = (float*)d_out;

  // ws layout: [0,256) counters (wctr[8], dctr[8], pad); [256, 256+512KB) double-buffered h (bf16)
  int* ctrs = (int*)d_ws;
  unsigned short* hbuf = (unsigned short*)((char*)d_ws + 256);

  init_counters_k<<<dim3(1), dim3(64), 0, stream>>>(ctrs);
  dim3 grid(NCC + 1, NBT);   // 17 x 8 = 136 persistent workgroups
  lstm_wmma_persistent<<<grid, dim3(THREADS), SMEM_BYTES, stream>>>(
      x, Wih, Whh, bih, bhh, Wd, bd, out, ctrs, hbuf);
}
